// DecoderAttention_22686017257615
// MI455X (gfx1250) — matrix operand order repaired and checked by reading, not compiled, not run on hardware
//
#include <hip/hip_runtime.h>
#include <hip/hip_bf16.h>
#include <math.h>

// ---------------------------------------------------------------------------
// Types for CDNA5 WMMA (gfx1250, wave32)
// ---------------------------------------------------------------------------
typedef __attribute__((ext_vector_type(16))) __bf16        v16bf;
typedef __attribute__((ext_vector_type(8)))  float         v8f;
typedef __attribute__((ext_vector_type(8)))  unsigned int  u8i;

// exact pointee type of the async-copy builtin (per hipcc diagnostic)
typedef int v4i_vs __attribute__((vector_size(16)));

#if defined(__has_builtin)
#if __has_builtin(__builtin_amdgcn_global_load_async_to_lds_b128)
#define ASYNC_COPY 1
#else
#define ASYNC_COPY 0
#endif
#else
#define ASYNC_COPY 0
#endif

__device__ __forceinline__ unsigned short f2bf(float f) {
    unsigned u = __float_as_uint(f);
    u = u + 0x7FFFu + ((u >> 16) & 1u);   // round-to-nearest-even
    return (unsigned short)(u >> 16);
}

__device__ __forceinline__ unsigned int pack_bf2(float a, float b) {
    return (unsigned int)f2bf(a) | ((unsigned int)f2bf(b) << 16);
}

__device__ __forceinline__ u8i make_u8(uint4 a, uint4 b) {
    u8i r;
    r[0] = a.x; r[1] = a.y; r[2] = a.z; r[3] = a.w;
    r[4] = b.x; r[5] = b.y; r[6] = b.z; r[7] = b.w;
    return r;
}

__device__ __forceinline__ v8f wmma_bf16(u8i a, u8i b, v8f c) {
    return __builtin_amdgcn_wmma_f32_16x16x32_bf16(
        false, __builtin_bit_cast(v16bf, a),
        false, __builtin_bit_cast(v16bf, b),
        (short)0, c, false, false);
}

// 128-bit global -> LDS copy: async (ASYNCcnt) on CDNA5, sync fallback
__device__ __forceinline__ void copy_b128(const unsigned short* g,
                                          unsigned short* l) {
#if ASYNC_COPY
    __builtin_amdgcn_global_load_async_to_lds_b128(
        (__attribute__((address_space(1))) v4i_vs*)g,
        (__attribute__((address_space(3))) v4i_vs*)l, 0, 0);
#else
    *(uint4*)l = *(const uint4*)g;
#endif
}

__device__ __forceinline__ void wait_async_all() {
#if ASYNC_COPY
    asm volatile("s_wait_asynccnt 0x0" ::: "memory");
#endif
}

// ---------------------------------------------------------------------------
// fp32 -> bf16 conversion (grid-stride)
// ---------------------------------------------------------------------------
__global__ void cvt_f32_to_bf16(const float* __restrict__ in,
                                unsigned short* __restrict__ out, int n) {
    for (int i = blockIdx.x * blockDim.x + threadIdx.x; i < n;
         i += gridDim.x * blockDim.x)
        out[i] = f2bf(in[i]);
}

// fp32 [K,N] -> bf16 transposed [N,K] (one-time weight reshape)
__global__ void cvt_f32_to_bf16_T(const float* __restrict__ in,
                                  unsigned short* __restrict__ out,
                                  int K, int N) {
    int total = K * N;
    for (int i = blockIdx.x * blockDim.x + threadIdx.x; i < total;
         i += gridDim.x * blockDim.x) {
        int k = i / N, n = i - k * N;
        out[(size_t)n * K + k] = f2bf(in[i]);
    }
}

// ---------------------------------------------------------------------------
// Tiled GEMM: C[M,N] = A[M,K](bf16, row-major) * Bt[N,K](bf16, row-major,
// i.e. B transposed) + bias[N].
// BM=BN=128, BK=32, 256 threads = 8 waves (4 along M x 2 along N).
// Double-buffered LDS staging via async global->LDS copies.
// Epilogues restructure the WMMA C layout through LDS so every global store
// is a coalesced B128.
// MODE 0: QKV split (Q,K row-major bf16 + V transposed bf16)
// MODE 1: fp32 row-major + bias (two 64-row passes through LDS)
// ---------------------------------------------------------------------------
#define GBM 128
#define GBN 128
#define GBK 32
#define LDS_STRIDE 40     // 32 + 8 pad (16B-aligned rows, de-skews banks)
#define TILE_U16  (GBM * LDS_STRIDE)   // 5120 u16 per staging buffer

template<int MODE>
__global__ __launch_bounds__(256)
void gemm_bf16_kernel(const unsigned short* __restrict__ A,
                      const unsigned short* __restrict__ Bt,
                      const float* __restrict__ bias,
                      unsigned short* __restrict__ qkOut,   // MODE 0
                      unsigned short* __restrict__ vtOut,   // MODE 0
                      float* __restrict__ fOut,             // MODE 1
                      int M, int N, int K) {
    // 40 KB shared pool: 4 staging buffers during the K loop,
    // reused as the output-restructuring buffer in the epilogue.
    // (buffer base addresses are computed arithmetically; pointer arrays of
    //  LDS addresses become unlinkable static initializers)
    __shared__ __attribute__((aligned(16))) unsigned short SH[4 * TILE_U16];

    const int tid  = threadIdx.x;
    const int lane = tid & 31;
    const int wave = tid >> 5;
    const int wm   = wave & 3;     // 0..3 : 32 rows each
    const int wn   = wave >> 2;    // 0..1 : 64 cols each
    const int lh   = lane >> 4;
    const int ll   = lane & 15;
    const int m0   = blockIdx.y * GBM;
    const int n0   = blockIdx.x * GBN;

    v8f acc[2][4];
#pragma unroll
    for (int mt = 0; mt < 2; ++mt)
#pragma unroll
        for (int nt = 0; nt < 4; ++nt)
            acc[mt][nt] = (v8f){0.f, 0.f, 0.f, 0.f, 0.f, 0.f, 0.f, 0.f};

    auto stage = [&](int buf, int k0) {
        unsigned short* Asb = SH + buf * TILE_U16;
        unsigned short* Bsb = SH + (2 + buf) * TILE_U16;
#pragma unroll
        for (int i = 0; i < 2; ++i) {
            int chunk = tid + i * 256;           // 0..511
            int row   = chunk >> 2;              // 0..127
            int kc    = (chunk & 3) * 8;         // 0,8,16,24
            copy_b128(A + (size_t)(m0 + row) * K + k0 + kc,
                      Asb + row * LDS_STRIDE + kc);
        }
#pragma unroll
        for (int i = 0; i < 2; ++i) {
            int chunk = tid + i * 256;
            int row   = chunk >> 2;
            int kc    = (chunk & 3) * 8;
            copy_b128(Bt + (size_t)(n0 + row) * K + k0 + kc,
                      Bsb + row * LDS_STRIDE + kc);
        }
    };

    stage(0, 0);
    for (int k0 = 0; k0 < K; k0 += GBK) {
        const int cur = (k0 >> 5) & 1;
        wait_async_all();       // this wave's async fills complete
        __syncthreads();        // all waves' fills visible
        if (k0 + GBK < K) stage(cur ^ 1, k0 + GBK);

        const unsigned short* Asb = SH + cur * TILE_U16;
        const unsigned short* Bsb = SH + (2 + cur) * TILE_U16;

        u8i afrag[2], bfrag[4];
#pragma unroll
        for (int mt = 0; mt < 2; ++mt) {
            const unsigned short* p =
                Asb + (wm * 32 + mt * 16 + ll) * LDS_STRIDE;
            uint4 lo = *(const uint4*)(p + lh * 8);        // K = lh*8..+7
            uint4 hi = *(const uint4*)(p + 16 + lh * 8);   // K = 16+lh*8..+7
            afrag[mt] = make_u8(lo, hi);
        }
#pragma unroll
        for (int nt = 0; nt < 4; ++nt) {
            const unsigned short* p =
                Bsb + (wn * 64 + nt * 16 + ll) * LDS_STRIDE + lh * 8;
            uint4 lo = *(const uint4*)(p);                 // K = lh*8..+7
            uint4 hi = *(const uint4*)(p + 16);            // K = 16+lh*8..+7
            bfrag[nt] = make_u8(lo, hi);
        }
#pragma unroll
        for (int mt = 0; mt < 2; ++mt)
#pragma unroll
            for (int nt = 0; nt < 4; ++nt)
                acc[mt][nt] = wmma_bf16(afrag[mt], bfrag[nt], acc[mt][nt]);
    }
    __syncthreads();   // staging buffers free for epilogue reuse

    if constexpr (MODE == 0) {
        // ---- bf16 restructure: SH as [128][136] u16, then coalesced B128 ---
        unsigned short* ST = SH;
#pragma unroll
        for (int mt = 0; mt < 2; ++mt)
#pragma unroll
            for (int nt = 0; nt < 4; ++nt) {
                int nl = wn * 64 + nt * 16 + ll;
                float bv = bias[n0 + nl];
#pragma unroll
                for (int r = 0; r < 8; ++r) {
                    int ml = wm * 32 + mt * 16 + r + lh * 8;
                    ST[ml * 136 + nl] = f2bf(acc[mt][nt][r] + bv);
                }
            }
        __syncthreads();
        if (n0 < 2048) {
            // Q,K: row-major [token, 2048]
#pragma unroll
            for (int i = 0; i < 8; ++i) {
                int c = tid + i * 256;          // 0..2047
                int row = c >> 4;               // 0..127
                int cc  = (c & 15) * 8;         // 0..120
                uint4 v = *(const uint4*)(&ST[row * 136 + cc]);
                *(uint4*)(&qkOut[(size_t)(m0 + row) * 2048 + n0 + cc]) = v;
            }
        } else {
            // V: transposed [b, h, hd, t]; stores coalesced along t
            int bb = m0 >> 10, t0 = m0 & 1023;
#pragma unroll
            for (int i = 0; i < 8; ++i) {
                int c  = tid + i * 256;
                int nl = c >> 4;                // 0..127 (hd within tile)
                int tc = (c & 15) * 8;          // 0..120 (t chunk)
                int hg = n0 - 2048 + nl;
                int h = hg >> 6, hd = hg & 63;
                unsigned int w[4];
#pragma unroll
                for (int j = 0; j < 4; ++j)
                    w[j] = (unsigned int)ST[(tc + 2 * j) * 136 + nl] |
                           ((unsigned int)ST[(tc + 2 * j + 1) * 136 + nl] << 16);
                uint4 v; v.x = w[0]; v.y = w[1]; v.z = w[2]; v.w = w[3];
                *(uint4*)(&vtOut[(((size_t)bb * 16 + h) * 64 + hd) * 1024 +
                                 t0 + tc]) = v;
            }
        }
    } else {
        // ---- fp32 out: two 64-row passes through LDS ([64][132] f32) ------
        float* SF = (float*)SH;
#pragma unroll
        for (int half = 0; half < 2; ++half) {
            __syncthreads();
            if ((wm >> 1) == half) {
#pragma unroll
                for (int mt = 0; mt < 2; ++mt)
#pragma unroll
                    for (int nt = 0; nt < 4; ++nt) {
                        int nl = wn * 64 + nt * 16 + ll;
                        float bv = bias[n0 + nl];
#pragma unroll
                        for (int r = 0; r < 8; ++r) {
                            int ml = (wm & 1) * 32 + mt * 16 + r + lh * 8;
                            SF[ml * 132 + nl] = acc[mt][nt][r] + bv;
                        }
                    }
            }
            __syncthreads();
#pragma unroll
            for (int i = 0; i < 8; ++i) {
                int c = tid + i * 256;          // 0..2047
                int row = c >> 5;               // 0..63
                int cc  = (c & 31) * 4;         // 0..124
                uint4 v = *(const uint4*)(&SF[row * 132 + cc]);
                *(uint4*)(&fOut[(size_t)(m0 + half * 64 + row) * N + n0 + cc]) = v;
            }
        }
    }
}

// ---------------------------------------------------------------------------
// Flash-style attention: one WG (128 thr, 4 waves) per (b, h, 64-query tile).
// Each wave independently owns 16 query rows; streams keys in 32-wide blocks
// with online softmax. Q/K fragments load straight from the row-major QK
// buffer, V fragments straight from the pre-transposed V^T buffer. Row sums
// of P come from an extra WMMA against an all-ones fragment (matrix pipe)
// instead of shuffle chains; only the row max uses lane reductions.
//   qk : bf16 [B*T, 2048]   (Q | K per token)
//   vt : bf16 [B, H, 64, 1024]
//   out: bf16 [B*T, 1024]
// ---------------------------------------------------------------------------
__global__ __launch_bounds__(128)
void attn_kernel(const unsigned short* __restrict__ qk,
                 const unsigned short* __restrict__ vt,
                 unsigned short* __restrict__ out) {
    __shared__ __attribute__((aligned(16))) float Pws[4][16 * 36];

    const int tid  = threadIdx.x;
    const int lane = tid & 31;
    const int wave = tid >> 5;
    const int lh   = lane >> 4;
    const int ll   = lane & 15;

    const int bid = blockIdx.x;
    const int b   = bid >> 8;             // / (H * 16)
    const int h   = (bid >> 4) & 15;
    const int qb  = bid & 15;

    const size_t tokBase = (size_t)b * 1024;
    const int q0   = qb * 64 + wave * 16;
    const int colQ = h * 64;
    const int colK = 1024 + h * 64;
    const unsigned short* vtb = vt + (((size_t)b * 16 + h) * 64) * 1024;
    const float scale = 0.125f;           // 1/sqrt(64)

    // constant all-ones bf16 B fragment (for P row sums via WMMA)
    u8i onesf;
#pragma unroll
    for (int i = 0; i < 8; ++i) onesf[i] = 0x3F803F80u;
    const v8f zero8 = (v8f){0.f, 0.f, 0.f, 0.f, 0.f, 0.f, 0.f, 0.f};

    // ---- Q fragments (A layout, K-dim = 64 -> 2 frags), loaded once ----
    u8i qf[2];
    {
        const unsigned short* qrow = qk + (tokBase + q0 + ll) * 2048 + colQ;
#pragma unroll
        for (int c = 0; c < 2; ++c) {
            uint4 lo = *(const uint4*)(qrow + c * 32 + lh * 8);
            uint4 hi = *(const uint4*)(qrow + c * 32 + 16 + lh * 8);
            qf[c] = make_u8(lo, hi);
        }
    }

    v8f o[4];
#pragma unroll
    for (int nt = 0; nt < 4; ++nt) o[nt] = zero8;
    float mrow[8], lrow[8];
#pragma unroll
    for (int r = 0; r < 8; ++r) { mrow[r] = -INFINITY; lrow[r] = 0.f; }

    for (int kb = 0; kb < 1024; kb += 32) {
        // prefetch next key block (global_prefetch_b8)
        if (kb + 32 < 1024)
            __builtin_prefetch(qk + (tokBase + kb + 32 + ll) * 2048 + colK, 0, 1);

        // ---- S = (Q K^T) * scale : two 16-key subtiles ----
        v8f s[2];
        s[0] = zero8; s[1] = zero8;
#pragma unroll
        for (int n = 0; n < 2; ++n) {
            const unsigned short* krow =
                qk + (tokBase + kb + n * 16 + ll) * 2048 + colK;
#pragma unroll
            for (int c = 0; c < 2; ++c) {
                // B fragment of K^T == A-style (table order) load of K's rows
                uint4 lo = *(const uint4*)(krow + c * 32 + lh * 8);
                uint4 hi = *(const uint4*)(krow + c * 32 + 16 + lh * 8);
                u8i kf = make_u8(lo, hi);
                s[n] = wmma_bf16(qf[c], kf, s[n]);
            }
        }
#pragma unroll
        for (int r = 0; r < 8; ++r) { s[0][r] *= scale; s[1][r] *= scale; }

        // ---- online softmax: row max via lane reduction, exp, rescale ----
        float p0[8], p1[8], cfa[8];
#pragma unroll
        for (int r = 0; r < 8; ++r) {
            float v = fmaxf(s[0][r], s[1][r]);
#pragma unroll
            for (int off = 1; off < 16; off <<= 1)
                v = fmaxf(v, __shfl_xor(v, off, 32));
            float mnew = fmaxf(mrow[r], v);
            float cf   = __expf(mrow[r] - mnew);
            p0[r] = __expf(s[0][r] - mnew);
            p1[r] = __expf(s[1][r] - mnew);
            mrow[r] = mnew;
            cfa[r]  = cf;
#pragma unroll
            for (int nt = 0; nt < 4; ++nt) o[nt][r] *= cf;
        }

        // ---- P: C layout -> wave-private LDS -> A layout (f32->bf16) ----
        float* P = &Pws[wave][0];
#pragma unroll
        for (int r = 0; r < 8; ++r) {
            int row = r + lh * 8;
            P[row * 36 + ll]      = p0[r];
            P[row * 36 + 16 + ll] = p1[r];
        }
        // same-wave DS ordering: compiler inserts s_wait_dscnt for the reload
        u8i pf;
        {
            const float* pr = &Pws[wave][ll * 36];
            float4 a  = *(const float4*)(pr + lh * 8);
            float4 bq = *(const float4*)(pr + lh * 8 + 4);
            float4 c  = *(const float4*)(pr + 16 + lh * 8);
            float4 d  = *(const float4*)(pr + 16 + lh * 8 + 4);
            pf[0] = pack_bf2(a.x, a.y);   pf[1] = pack_bf2(a.z, a.w);
            pf[2] = pack_bf2(bq.x, bq.y); pf[3] = pack_bf2(bq.z, bq.w);
            pf[4] = pack_bf2(c.x, c.y);   pf[5] = pack_bf2(c.z, c.w);
            pf[6] = pack_bf2(d.x, d.y);   pf[7] = pack_bf2(d.z, d.w);
        }

        // ---- row sums via matrix pipe: ls = P @ ones (same layout as O) ---
        v8f ls = wmma_bf16(pf, onesf, zero8);
#pragma unroll
        for (int r = 0; r < 8; ++r) lrow[r] = lrow[r] * cfa[r] + ls[r];

        // ---- O += P @ V : 4 hd chunks, V fragments direct from global V^T --
#pragma unroll
        for (int nt = 0; nt < 4; ++nt) {
            const unsigned short* vr =
                vtb + (size_t)(nt * 16 + ll) * 1024 + kb + lh * 8;
            uint4 lo = *(const uint4*)(vr);
            uint4 hi = *(const uint4*)(vr + 16);
            u8i vf = make_u8(lo, hi);
            o[nt] = wmma_bf16(pf, vf, o[nt]);
        }
    }

    // ---- normalize; restructure through wave-private LDS; coalesced store --
    float inv[8];
#pragma unroll
    for (int r = 0; r < 8; ++r) inv[r] = 1.0f / lrow[r];
    unsigned short* OS = (unsigned short*)&Pws[wave][0];   // [16][72] u16
#pragma unroll
    for (int nt = 0; nt < 4; ++nt)
#pragma unroll
        for (int r = 0; r < 8; ++r)
            OS[(r + lh * 8) * 72 + nt * 16 + ll] = f2bf(o[nt][r] * inv[r]);
    {
        int row = lane >> 1, halfc = (lane & 1) * 32;
        const unsigned short* src = &OS[row * 72 + halfc];
        unsigned short* dst =
            out + (tokBase + q0 + row) * 1024 + h * 64 + halfc;
#pragma unroll
        for (int j = 0; j < 4; ++j)
            *(uint4*)(dst + j * 8) = *(const uint4*)(src + j * 8);
    }
}

// ---------------------------------------------------------------------------
// Launch
// ---------------------------------------------------------------------------
extern "C" void kernel_launch(void* const* d_in, const int* in_sizes, int n_in,
                              void* d_out, int out_size, void* d_ws, size_t ws_size,
                              hipStream_t stream) {
    (void)in_sizes; (void)n_in; (void)out_size; (void)ws_size;

    const float* hidden = (const float*)d_in[0];   // [16, 1024, 1024]
    const float* w_qkv  = (const float*)d_in[1];   // [1024, 3072]
    const float* b_qkv  = (const float*)d_in[2];   // [3072]
    const float* w_out  = (const float*)d_in[3];   // [1024, 1024]
    const float* b_out  = (const float*)d_in[4];   // [1024]
    float* outp = (float*)d_out;                   // [16*1024, 1024]

    const int M = 16384, D = 1024, N3 = 3072;

    // workspace layout (bytes)
    char* ws = (char*)d_ws;
    unsigned short* hid_bf  = (unsigned short*)(ws);                 // 33,554,432
    unsigned short* wqkvT   = (unsigned short*)(ws + 33554432);      //  6,291,456
    unsigned short* woutT   = (unsigned short*)(ws + 39845888);      //  2,097,152
    unsigned short* qk_bf   = (unsigned short*)(ws + 41943040);      // 67,108,864
    unsigned short* vt_bf   = (unsigned short*)(ws + 109051904);     // 33,554,432
    unsigned short* attn_bf = (unsigned short*)(ws + 142606336);     // 33,554,432

    // 1) fp32 -> bf16 staging; weights transposed once so all GEMM staging
    //    and WMMA fragment loads are contiguous 128-bit accesses
    cvt_f32_to_bf16<<<2048, 256, 0, stream>>>(hidden, hid_bf, M * D);
    cvt_f32_to_bf16_T<<<2048, 256, 0, stream>>>(w_qkv, wqkvT, D, N3);
    cvt_f32_to_bf16_T<<<1024, 256, 0, stream>>>(w_out, woutT, D, D);

    // 2) QKV projection: Q,K -> row-major buffer; V -> transposed buffer
    dim3 g1(N3 / GBN, M / GBM);
    gemm_bf16_kernel<0><<<g1, 256, 0, stream>>>(
        hid_bf, wqkvT, b_qkv, qk_bf, vt_bf, nullptr, M, N3, D);

    // 3) attention: grid = B * H * (T/64) = 16*16*16
    attn_kernel<<<4096, 128, 0, stream>>>(qk_bf, vt_bf, attn_bf);

    // 4) output projection -> fp32 d_out (+bias)
    dim3 g2(D / GBN, M / GBM);
    gemm_bf16_kernel<1><<<g2, 256, 0, stream>>>(
        attn_bf, woutT, b_out, nullptr, nullptr, outp, M, D, D);
}
